// VectorTokenizer_23596550324863
// MI455X (gfx1250) — matrix operand order repaired and checked by reading, not compiled, not run on hardware
//
#include <hip/hip_runtime.h>
#include <hip/hip_bf16.h>

typedef __attribute__((ext_vector_type(16))) __bf16       v16bf;
typedef __attribute__((ext_vector_type(8)))  float        v8f;
typedef __attribute__((ext_vector_type(4)))  unsigned int u32x4;

union FragBF { v16bf v; u32x4 u[2]; };

__device__ __forceinline__ unsigned short f2bf(float f) {
    unsigned x = __float_as_uint(f);
    unsigned r = x + 0x7FFFu + ((x >> 16) & 1u);   // round-to-nearest-even
    return (unsigned short)(r >> 16);
}

// Async global->LDS 16-byte copy (CDNA5, ASYNCcnt-tracked, no VGPR staging)
__device__ __forceinline__ void async_cp_b128(unsigned ldsOff, const void* gaddr) {
    asm volatile("global_load_async_to_lds_b128 %0, %1, off"
                 :: "v"(ldsOff), "v"((unsigned long long)(uintptr_t)gaddr)
                 : "memory");
}

// ---------------------------------------------------------------------------
// f32 -> bf16 elementwise
// ---------------------------------------------------------------------------
__global__ void f32_to_bf16_kernel(const float* __restrict__ in,
                                   unsigned short* __restrict__ out, int n) {
    int i = blockIdx.x * 256 + threadIdx.x;
    if (i < n) out[i] = f2bf(in[i]);
}

// ---------------------------------------------------------------------------
// W (K x N, f32, row-major) -> Wt (N x K, bf16, row-major), coalesced both ways
// ---------------------------------------------------------------------------
__global__ __launch_bounds__(256) void transpose_f32_to_bf16(
    const float* __restrict__ W, unsigned short* __restrict__ Wt, int K, int N)
{
    __shared__ __attribute__((aligned(16))) unsigned short tile[32][33];
    const int nBase = blockIdx.x * 32;
    const int kBase = blockIdx.y * 32;
    const int tx = threadIdx.x, ty = threadIdx.y;
#pragma unroll
    for (int i = 0; i < 4; ++i) {
        int r = ty + i * 8;
        tile[r][tx] = f2bf(W[(size_t)(kBase + r) * N + nBase + tx]);
    }
    __syncthreads();
#pragma unroll
    for (int i = 0; i < 4; ++i) {
        int r = ty + i * 8;
        Wt[(size_t)(nBase + r) * K + kBase + tx] = tile[tx][r];
    }
}

// ---------------------------------------------------------------------------
// bf16 WMMA GEMM:  C(MxN) = A(MxK) @ B(KxN) + bias, optional relu.
// A: M x K bf16 row-major. Bt: N x K bf16 row-major (B pre-transposed).
// WG tile 64(M) x 256(N) x 32(K); 8 waves as 2x4, each wave 32x64 = 2x4 WMMA.
// Double-buffered LDS filled by async global->LDS copies (ASYNCcnt pipeline).
// ---------------------------------------------------------------------------
#define TM 64
#define TN 256
#define TK 32
#define LDS_S 40                     // padded LDS row stride in halfs (80 B)
#define A_HALFS (TM * LDS_S)         // 2560 halfs / buffer
#define B_HALFS (TN * LDS_S)         // 10240 halfs / buffer

__global__ __launch_bounds__(256) void gemm_bf16_wmma(
    const unsigned short* __restrict__ A,
    const unsigned short* __restrict__ Bt,
    const float* __restrict__ bias,
    float* __restrict__ outF,            // may be null
    unsigned short* __restrict__ outBF,  // may be null
    int M, int N, int K, int relu)
{
    __shared__ __attribute__((aligned(16))) unsigned short As[2][A_HALFS];
    __shared__ __attribute__((aligned(16))) unsigned short Bs[2][B_HALFS];

    const int t    = threadIdx.x;
    const int lane = t & 31;
    const int wid  = t >> 5;
    const int wm   = wid >> 2;        // 0..1 : wave M index
    const int wn   = wid & 3;         // 0..3 : wave N index
    const int lm   = lane & 15;
    const int lh   = lane >> 4;

    const int m0 = blockIdx.y * TM;
    const int n0 = blockIdx.x * TN;

    const int arow = t >> 2;          // 0..63
    const int acol = (t & 3) * 8;     // halfs

    const unsigned asBase = (unsigned)(uintptr_t)&As[0][0];
    const unsigned bsBase = (unsigned)(uintptr_t)&Bs[0][0];

    v8f acc[2][4] = {};

    // stage one K-tile into LDS buffer `buf` via async copies (5 per thread)
    auto stage = [&](int buf, int k0) {
        async_cp_b128(asBase + (unsigned)(buf * A_HALFS + arow * LDS_S + acol) * 2u,
                      A + (size_t)(m0 + arow) * K + k0 + acol);
#pragma unroll
        for (int c = 0; c < 4; ++c) {
            const int t2   = t + c * 256;
            const int brow = t2 >> 2;          // 0..255
            const int bcol = (t2 & 3) * 8;
            async_cp_b128(bsBase + (unsigned)(buf * B_HALFS + brow * LDS_S + bcol) * 2u,
                          Bt + (size_t)(n0 + brow) * K + k0 + bcol);
        }
    };

    const int ktiles = K / TK;
    stage(0, 0);                                   // prologue

    for (int kt = 0; kt < ktiles; ++kt) {
        const int cb = kt & 1;
        if (kt + 1 < ktiles) {
            stage(cb ^ 1, (kt + 1) * TK);          // overlap next stage w/ compute
            asm volatile("s_wait_asynccnt 0x5" ::: "memory");  // stage kt done
        } else {
            asm volatile("s_wait_asynccnt 0x0" ::: "memory");
        }
        __syncthreads();                           // all waves' tiles visible

        FragBF a[2], b[4];
#pragma unroll
        for (int mi = 0; mi < 2; ++mi) {
            const unsigned short* p = &As[cb][(wm * 32 + mi * 16 + lm) * LDS_S];
            a[mi].u[0] = *(const u32x4*)(p + lh * 8);        // K 0..7  / 8..15
            a[mi].u[1] = *(const u32x4*)(p + 16 + lh * 8);   // K 16..23/ 24..31
        }
#pragma unroll
        for (int ni = 0; ni < 4; ++ni) {
            const unsigned short* p = &Bs[cb][(wn * 64 + ni * 16 + lm) * LDS_S];
            b[ni].u[0] = *(const u32x4*)(p + lh * 8);        // K 0..7  / 8..15
            b[ni].u[1] = *(const u32x4*)(p + 16 + lh * 8);   // K 16..23/ 24..31
        }
#pragma unroll
        for (int mi = 0; mi < 2; ++mi)
#pragma unroll
            for (int ni = 0; ni < 4; ++ni)
                acc[mi][ni] = __builtin_amdgcn_wmma_f32_16x16x32_bf16(
                    false, a[mi].v, false, b[ni].v,
                    (short)0, acc[mi][ni], false, false);

        __syncthreads();            // protect buffer before it is re-staged
    }

    // epilogue: C layout VGPR r -> M = r (lanes 0-15) or r+8 (lanes 16-31)
    const int cm = lh * 8;
    const int cn = lm;
#pragma unroll
    for (int mi = 0; mi < 2; ++mi) {
#pragma unroll
        for (int ni = 0; ni < 4; ++ni) {
            const int   gn = n0 + wn * 64 + ni * 16 + cn;
            const float bv = bias ? bias[gn] : 0.0f;
#pragma unroll
            for (int r = 0; r < 8; ++r) {
                const int gm = m0 + wm * 32 + mi * 16 + cm + r;
                float v = acc[mi][ni][r] + bv;
                if (relu) v = v > 0.0f ? v : 0.0f;
                const size_t idx = (size_t)gm * N + gn;
                if (outF)  outF[idx]  = v;
                if (outBF) outBF[idx] = f2bf(v);
            }
        }
    }
}

// ---------------------------------------------------------------------------
// Per-row tokenization + codebook gather:
//   tokens[k] = argmax_v (z[row,k,v] + Gumbel(hash))     (deterministic)
//   z_q[row,:] = sum_k codebook[k*512 + tokens[k], :]
// ---------------------------------------------------------------------------
__global__ __launch_bounds__(256) void tokenize_kernel(
    const float* __restrict__ z,          // 4096 x 16384
    const float* __restrict__ codebook,   // 16384 x 256
    float* __restrict__ zqF,              // 4096 x 256
    unsigned short* __restrict__ zqBF)
{
    __shared__ float sval[256];
    __shared__ int   sidx[256];
    const int row = blockIdx.x;
    const int t   = threadIdx.x;
    float acc = 0.0f;

    for (int k = 0; k < 32; ++k) {
        const float* logits = z + (size_t)row * 16384 + k * 512;
        float bv = -1e30f; int bi = 0;
#pragma unroll
        for (int c = 0; c < 2; ++c) {
            const int v = t + c * 256;
            unsigned s = (unsigned)(row * 16384 + k * 512 + v);
            s = s * 2654435761u + 12345u;
            s ^= s >> 16; s *= 0x85ebca6bu; s ^= s >> 13; s *= 0xc2b2ae35u; s ^= s >> 16;
            const float u  = (float)(s & 0xFFFFFFu) * (1.0f / 16777216.0f) + 1e-7f;
            const float g  = -__logf(-__logf(u));      // Gumbel(0,1)
            const float sc = logits[v] + g;
            if (sc > bv) { bv = sc; bi = v; }
        }
        __syncthreads();                 // protect LDS from previous iteration
        sval[t] = bv; sidx[t] = bi;
        __syncthreads();
        for (int s = 128; s > 0; s >>= 1) {
            if (t < s && sval[t + s] > sval[t]) {
                sval[t] = sval[t + s]; sidx[t] = sidx[t + s];
            }
            __syncthreads();
        }
        const int tok = k * 512 + sidx[0];
        acc += codebook[(size_t)tok * 256 + t];
    }
    zqF[(size_t)row * 256 + t]  = acc;
    zqBF[(size_t)row * 256 + t] = f2bf(acc);
}

// ---------------------------------------------------------------------------
extern "C" void kernel_launch(void* const* d_in, const int* in_sizes, int n_in,
                              void* d_out, int out_size, void* d_ws, size_t ws_size,
                              hipStream_t stream)
{
    (void)in_sizes; (void)n_in; (void)out_size; (void)ws_size;
    const float* x        = (const float*)d_in[0];
    const float* enc_w1   = (const float*)d_in[1];
    const float* enc_b1   = (const float*)d_in[2];
    const float* enc_w2   = (const float*)d_in[3];
    const float* enc_b2   = (const float*)d_in[4];
    const float* codebook = (const float*)d_in[5];
    const float* dec_w1   = (const float*)d_in[6];
    const float* dec_b1   = (const float*)d_in[7];
    const float* dec_w2   = (const float*)d_in[8];
    const float* dec_b2   = (const float*)d_in[9];

    const int B = 4096, D = 512, E = 256, Z = 16384;

    float* z   = (float*)d_out;              // B*Z
    float* zq  = z  + (size_t)B * Z;         // B*E
    float* rec = zq + (size_t)B * E;         // B*D

    char* wsb = (char*)d_ws;
    size_t off = 0;
    auto alloc = [&](size_t bytes) -> void* {
        void* p = wsb + off;
        off = (off + bytes + 255) & ~(size_t)255;
        return p;
    };
    unsigned short* x_bf  = (unsigned short*)alloc((size_t)B * D * 2);
    unsigned short* h_bf  = (unsigned short*)alloc((size_t)B * D * 2);
    unsigned short* w1t   = (unsigned short*)alloc((size_t)D * D * 2);
    unsigned short* w2t   = (unsigned short*)alloc((size_t)Z * D * 2);
    unsigned short* d1t   = (unsigned short*)alloc((size_t)D * E * 2);
    unsigned short* d2t   = (unsigned short*)alloc((size_t)D * D * 2);
    unsigned short* zq_bf = (unsigned short*)alloc((size_t)B * E * 2);
    unsigned short* hd_bf = (unsigned short*)alloc((size_t)B * D * 2);

    // precision conversion / weight transposes (per-launch, deterministic)
    f32_to_bf16_kernel<<<(B * D) / 256, 256, 0, stream>>>(x, x_bf, B * D);
    transpose_f32_to_bf16<<<dim3(D / 32, D / 32), dim3(32, 8), 0, stream>>>(enc_w1, w1t, D, D);
    transpose_f32_to_bf16<<<dim3(Z / 32, D / 32), dim3(32, 8), 0, stream>>>(enc_w2, w2t, D, Z);
    transpose_f32_to_bf16<<<dim3(D / 32, E / 32), dim3(32, 8), 0, stream>>>(dec_w1, d1t, E, D);
    transpose_f32_to_bf16<<<dim3(D / 32, D / 32), dim3(32, 8), 0, stream>>>(dec_w2, d2t, D, D);

    // encoder: h = relu(x @ W1 + b1)   [bf16 out only]
    gemm_bf16_wmma<<<dim3(D / TN, B / TM), 256, 0, stream>>>(
        x_bf, w1t, enc_b1, nullptr, h_bf, B, D, D, 1);
    // encoder: z = h @ W2 + b2         [f32 out to d_out]
    gemm_bf16_wmma<<<dim3(Z / TN, B / TM), 256, 0, stream>>>(
        h_bf, w2t, enc_b2, z, nullptr, B, Z, D, 0);
    // tokens + z_q
    tokenize_kernel<<<B, 256, 0, stream>>>(z, codebook, zq, zq_bf);
    // decoder: hd = relu(z_q @ dW1 + db1)
    gemm_bf16_wmma<<<dim3(D / TN, B / TM), 256, 0, stream>>>(
        zq_bf, d1t, dec_b1, nullptr, hd_bf, B, D, E, 1);
    // decoder: rec = hd @ dW2 + db2
    gemm_bf16_wmma<<<dim3(D / TN, B / TM), 256, 0, stream>>>(
        hd_bf, d2t, dec_b2, rec, nullptr, B, D, D, 0);
}
